// LTF_28698971472230
// MI455X (gfx1250) — matrix operand order repaired and checked by reading, not compiled, not run on hardware
//
#include <hip/hip_runtime.h>
#include <math.h>

// ---------------------------------------------------------------------------
// Problem constants (match reference)
// ---------------------------------------------------------------------------
constexpr int NROWS   = 4 * 4096;       // B*S = 16384
constexpr int DIMOUT  = 256;
constexpr int DFC     = 128;            // fourier dims
constexpr int DSC     = 128;            // spline dims
constexpr int GFC     = 5;              // harmonics
constexpr int KF      = DFC * GFC * 2;  // 1280 fourier features
constexpr int KS      = DSC + DSC * 8;  // 1152 spline features (tanh + 8 bspline)
constexpr float MINFRQ = 1e-6f;
constexpr float LNEPS  = 1e-5f;
constexpr float PMIX   = 0.5f;

// Tiling: 64 rows/block -> 256 blocks; ~121KB LDS -> 2 blocks per WGP fit.
constexpr int TM      = 64;             // rows per workgroup
constexpr int THREADS = 256;            // 8 wave32s
constexpr int CHUNK   = 128;            // K-chunk staged in LDS
constexpr int NCHF    = KF / CHUNK;     // 10
constexpr int NCHS    = KS / CHUNK;     // 9
constexpr int FSTR    = CHUNK + 8;      // f16 row stride (16B aligned, bank-skewed)
constexpr int OSTR    = DIMOUT + 4;     // f32 out row stride (bank-skewed)

typedef _Float16 h8  __attribute__((ext_vector_type(8)));
typedef _Float16 h16 __attribute__((ext_vector_type(16)));
typedef float    v8f __attribute__((ext_vector_type(8)));
typedef int      v4i __attribute__((vector_size(4 * sizeof(int))));  // matches builtin

union H16 { h16 v; h8 h[2]; };

static __device__ inline v8f zero_v8f() {
    v8f z;
#pragma unroll
    for (int i = 0; i < 8; ++i) z[i] = 0.0f;
    return z;
}

// ---------------------------------------------------------------------------
// CDNA5 async global->LDS staging (ASYNCcnt path), with portable fallback.
// Builtin signature (from hipcc diagnostic): (global v4i*, local v4i*, imm, imm)
// ---------------------------------------------------------------------------
#if __has_builtin(__builtin_amdgcn_global_load_async_to_lds_b128)
#define LTF_HAS_ASYNC 1
#else
#define LTF_HAS_ASYNC 0
#endif

static __device__ inline void stage16B(const _Float16* __restrict__ src,
                                       _Float16* __restrict__ dst) {
#if LTF_HAS_ASYNC
    __builtin_amdgcn_global_load_async_to_lds_b128(
        (__attribute__((address_space(1))) v4i*)src,
        (__attribute__((address_space(3))) v4i*)dst, 0, 0);
#else
    *(h8*)dst = *(const h8*)src;
#endif
}

static __device__ inline void wait_async_done() {
#if LTF_HAS_ASYNC
#if __has_builtin(__builtin_amdgcn_s_wait_asynccnt)
    __builtin_amdgcn_s_wait_asynccnt(0);
#else
    asm volatile("s_wait_asynccnt 0" ::: "memory");
#endif
#endif
}

// Cubic B-spline bases on the reference's uniform extended grid:
// grid[j] = j*0.4 - 2.2, j = 0..11 ; Cox–de Boor with K=3 -> 8 bases.
static __device__ inline void bspline8(float x, float* __restrict__ out) {
    const float h = 0.4f;
    float g[12];
#pragma unroll
    for (int j = 0; j < 12; ++j) g[j] = (float)j * h - 2.2f;
    float b[11];
#pragma unroll
    for (int j = 0; j < 11; ++j) b[j] = (x >= g[j] && x < g[j + 1]) ? 1.0f : 0.0f;
    const float i1 = 1.0f / h, i2 = 1.0f / (2.0f * h), i3 = 1.0f / (3.0f * h);
#pragma unroll
    for (int j = 0; j < 10; ++j)
        b[j] = (x - g[j]) * i1 * b[j] + (g[j + 2] - x) * i1 * b[j + 1];
#pragma unroll
    for (int j = 0; j < 9; ++j)
        b[j] = (x - g[j]) * i2 * b[j] + (g[j + 3] - x) * i2 * b[j + 1];
#pragma unroll
    for (int j = 0; j < 8; ++j)
        b[j] = (x - g[j]) * i3 * b[j] + (g[j + 4] - x) * i3 * b[j + 1];
#pragma unroll
    for (int j = 0; j < 8; ++j) out[j] = b[j];
}

// ---------------------------------------------------------------------------
// Prep: weights to f16, row-major [out][k] so fragments are contiguous runs.
// ---------------------------------------------------------------------------
__global__ void ltf_prep_weights(const float* __restrict__ fw,
                                 const float* __restrict__ bw,
                                 const float* __restrict__ sw,
                                 _Float16* __restrict__ WF,
                                 _Float16* __restrict__ WS) {
    const int totF = DFC * KF;
    const int totS = DSC * KS;
    for (int idx = blockIdx.x * blockDim.x + threadIdx.x; idx < totF + totS;
         idx += gridDim.x * blockDim.x) {
        if (idx < totF) {
            int o = idx / KF, k = idx % KF;
            float v = (k < 640) ? fw[o * 640 + k] : fw[DFC * 640 + o * 640 + (k - 640)];
            WF[o * KF + k] = (_Float16)v;
        } else {
            int j = idx - totF;
            int o = j / KS, k = j % KS;
            float v = (k < DSC) ? bw[o * DSC + k] : sw[o * 1024 + (k - DSC)];
            WS[o * KS + k] = (_Float16)v;
        }
    }
}

// ---------------------------------------------------------------------------
// Main fused kernel
// ---------------------------------------------------------------------------
__global__ void __launch_bounds__(THREADS)
ltf_kan_fused(const float* __restrict__ ts,
              const float* __restrict__ freq_theta,
              const float* __restrict__ bias_fourier,
              const float* __restrict__ w1w,
              const float* __restrict__ w1b,
              const float* __restrict__ ln_w,
              const float* __restrict__ ln_b,
              const float* __restrict__ sc_w,
              const _Float16* __restrict__ WF,
              const _Float16* __restrict__ WS,
              float* __restrict__ out) {
    extern __shared__ __align__(16) char smem[];
    float*    sOut  = (float*)smem;                    // TM * OSTR f32
    _Float16* sFeat = (_Float16*)(sOut + TM * OSTR);   // TM * FSTR f16
    _Float16* sW    = sFeat + TM * FSTR;               // 128 * FSTR f16 (weight chunk)
    float*    sFW   = (float*)(sW + DFC * FSTR);       // 128
    float*    sFB   = sFW + DFC;                       // 128
    float*    sW1   = sFB + DFC;                       // 128
    float*    sB1   = sW1 + DSC;                       // 128
    float*    sT    = sB1 + DSC;                       // TM timestamps

    const int tid  = threadIdx.x;
    const int wave = tid >> 5;
    const int lane = tid & 31;
    const int n0   = blockIdx.x * TM;

    // --- per-block constants ---------------------------------------------
    if (tid < DFC) {
        int d = tid;
        float bf = __expf((-9.0f * (float)d / 127.0f) * 2.302585093f); // 10^(-9d/127)
        float th = freq_theta[d];
        float sp = (th > 20.0f) ? th : log1pf(__expf(th));             // softplus
        sFW[d] = bf * (sp + MINFRQ);
        sFB[d] = bias_fourier[d];
        sW1[d] = w1w[d];
        sB1[d] = w1b[d];
    }
    if (tid < TM) sT[tid] = ts[n0 + tid];
    __syncthreads();

    // --- wave -> tile mapping (4 row tiles x 8 col tiles, 4 tiles/wave) ---
    const int rt    = wave & 3;            // row tile 0..3 (16 rows each)
    const int ct0   = (wave >> 2) * 4;     // col tiles ct0..ct0+3
    const int rowA  = rt * 16 + (lane & 15);
    const int koffA = (lane < 16) ? 0 : 8;   // A fragment K offset (ISA layout)
    const int koffB = (lane < 16) ? 0 : 8;   // B fragment K offset (ISA layout)
    const int colB  = lane & 15;
    const int rowC  = rt * 16 + ((lane < 16) ? 0 : 8);
    const int colC  = lane & 15;

    // ===================== two GEMM phases ================================
#pragma unroll 1
    for (int phase = 0; phase < 2; ++phase) {
        const _Float16* Wmat = (phase == 0) ? WF : WS;
        const int Kdim  = (phase == 0) ? KF : KS;
        const int nch   = (phase == 0) ? NCHF : NCHS;
        const int cbase = (phase == 0) ? 0 : 128;

        v8f acc[4];
#pragma unroll
        for (int c = 0; c < 4; ++c) acc[c] = zero_v8f();

#pragma unroll 1
        for (int ch = 0; ch < nch; ++ch) {
            // ---- issue async DMA of this chunk's weights (overlaps VALU) --
            // 128 cols x 128 k (f16) -> sW, padded stride FSTR
            for (int it = tid; it < DFC * (CHUNK / 8); it += THREADS) {
                int col  = it >> 4;          // /16 parts per col
                int part = it & 15;
                stage16B(Wmat + (size_t)col * Kdim + ch * CHUNK + part * 8,
                         &sW[col * FSTR + part * 8]);
            }

            // ---- compute this chunk's features into sFeat ----------------
            if (phase == 0) {
                const bool isCos = (ch < 5);
                const int kbase = (ch % 5) * CHUNK;
                for (int it = tid; it < TM * CHUNK; it += THREADS) {
                    int r  = it >> 7;
                    int kk = it & 127;
                    int kd = kbase + kk;          // d*5 + g
                    int d  = kd / GFC;
                    int g  = kd - d * GFC;
                    float xf  = fmaf(sT[r], sFW[d], sFB[d]);
                    float ang = xf * (float)(g + 1);
                    float v   = isCos ? __cosf(ang) : __sinf(ang);
                    sFeat[r * FSTR + kk] = (_Float16)v;
                }
            } else if (ch == 0) {
                for (int it = tid; it < TM * CHUNK; it += THREADS) {
                    int r = it >> 7;
                    int d = it & 127;
                    float x = fmaf(sT[r], sW1[d], sB1[d]);
                    sFeat[r * FSTR + d] = (_Float16)tanhf(x);
                }
            } else {
                const int d0 = (ch - 1) * 16;     // 16 dims/chunk, 8 bases each
                for (int it = tid; it < TM * 16; it += THREADS) {
                    int r  = it >> 4;
                    int dl = it & 15;
                    float x = fmaf(sT[r], sW1[d0 + dl], sB1[d0 + dl]);
                    float bs[8];
                    bspline8(x, bs);
                    H16 pack;
#pragma unroll
                    for (int j = 0; j < 8; ++j) pack.h[0][j] = (_Float16)bs[j];
                    *(h8*)&sFeat[r * FSTR + dl * 8] = pack.h[0];
                }
            }
            wait_async_done();
            __syncthreads();

            // ---- WMMAs: A from sFeat, B from staged sW -------------------
            H16 A[4];
#pragma unroll
            for (int ks = 0; ks < 4; ++ks) {
                const h8* p = (const h8*)&sFeat[rowA * FSTR + ks * 32 + koffA];
                A[ks].h[0] = p[0];   // K = koffA + 0..7
                A[ks].h[1] = p[2];   // K = koffA + 16..23
            }
#pragma unroll
            for (int c = 0; c < 4; ++c) {
                int col = (ct0 + c) * 16 + colB;
                const _Float16* wrow = &sW[col * FSTR + koffB];
#pragma unroll
                for (int ks = 0; ks < 4; ++ks) {
                    H16 Bm;
                    const h8* wp = (const h8*)(wrow + ks * 32);
                    Bm.h[0] = wp[0];
                    Bm.h[1] = wp[2];
                    acc[c] = __builtin_amdgcn_wmma_f32_16x16x32_f16(
                        false, A[ks].v, false, Bm.v, (short)0, acc[c], false, false);
                }
            }
            __syncthreads();
        }

        // stage accumulators to sOut (phase 0 -> cols 0..127, phase 1 -> 128..255)
#pragma unroll
        for (int c = 0; c < 4; ++c) {
            int cb = cbase + (ct0 + c) * 16 + colC;
#pragma unroll
            for (int j = 0; j < 8; ++j)
                sOut[(rowC + j) * OSTR + cb] = acc[c][j];
        }
    }
    __syncthreads();

    // ===================== LayerNorm epilogue =============================
    for (int r = wave * (TM / 8); r < (wave + 1) * (TM / 8); ++r) {
        float vals[8];
        float s = 0.0f, sq = 0.0f;
#pragma unroll
        for (int i = 0; i < 8; ++i) {
            int c = lane + 32 * i;
            float mix = (c < 128) ? PMIX : (1.0f - PMIX);
            float v = sOut[r * OSTR + c] * mix;
            vals[i] = v;
            s  += v;
            sq += v * v;
        }
#pragma unroll
        for (int off = 16; off > 0; off >>= 1) {
            s  += __shfl_xor(s, off, 32);
            sq += __shfl_xor(sq, off, 32);
        }
        float mu  = s * (1.0f / 256.0f);
        float var = sq * (1.0f / 256.0f) - mu * mu;
        float rs  = rsqrtf(var + LNEPS);
        size_t nbase = (size_t)(n0 + r) * DIMOUT;
#pragma unroll
        for (int i = 0; i < 8; ++i) {
            int c = lane + 32 * i;
            float o = (vals[i] - mu) * rs * ln_w[c] + ln_b[c];
            out[nbase + c] = o * sc_w[c];
        }
    }
}

// ---------------------------------------------------------------------------
// Launch
// ---------------------------------------------------------------------------
extern "C" void kernel_launch(void* const* d_in, const int* in_sizes, int n_in,
                              void* d_out, int out_size, void* d_ws, size_t ws_size,
                              hipStream_t stream) {
    const float* timestamps     = (const float*)d_in[0];
    const float* freq_theta     = (const float*)d_in[1];
    const float* bias_fourier   = (const float*)d_in[2];
    const float* fourier_weight = (const float*)d_in[3];
    const float* w1_spline_w    = (const float*)d_in[4];
    const float* w1_spline_b    = (const float*)d_in[5];
    const float* base_weight    = (const float*)d_in[6];
    const float* spline_weight  = (const float*)d_in[7];
    const float* scale_weight   = (const float*)d_in[8];
    const float* ln_weight      = (const float*)d_in[9];
    const float* ln_bias        = (const float*)d_in[10];
    float* outp = (float*)d_out;

    _Float16* WF = (_Float16*)d_ws;                    // 128 x 1280 f16
    _Float16* WS = WF + (size_t)DFC * KF;              // 128 x 1152 f16

    ltf_prep_weights<<<512, 256, 0, stream>>>(fourier_weight, base_weight,
                                              spline_weight, WF, WS);

    size_t shmem = (size_t)TM * OSTR * sizeof(float)            // sOut
                 + (size_t)TM * FSTR * sizeof(_Float16)         // sFeat
                 + (size_t)DFC * FSTR * sizeof(_Float16)        // sW chunk
                 + (4 * 128 + TM) * sizeof(float);              // consts
    ltf_kan_fused<<<NROWS / TM, THREADS, shmem, stream>>>(
        timestamps, freq_theta, bias_fourier, w1_spline_w, w1_spline_b,
        ln_weight, ln_bias, scale_weight, WF, WS, outp);
}
